// My_GraphCON_GAT_51316269252843
// MI455X (gfx1250) — matrix operand order repaired and checked by reading, not compiled, not run on hardware
//
#include <hip/hip_runtime.h>
#include <hip/hip_bf16.h>

#define N_NODES 20000
#define N_EDGES 320000
#define BATCH   2
#define DIN     32
#define HID     64
#define HEADS   4
#define DOUT    8
#define MROWS   (N_NODES * BATCH)   /* 40000, multiple of 16 */
#define SLOPE   0.2f

typedef __attribute__((ext_vector_type(16))) _Float16 v16h;
typedef __attribute__((ext_vector_type(8)))  _Float16 v8h;
typedef __attribute__((ext_vector_type(8)))  float    v8f;

// ---------------- WMMA helpers (CDNA5 16x16x32 f16, f32 acc) ----------------

// A tile 16x32 (MxK) from f16 row-major [rows x ldk].
// lanes 0-15 row=lane, K = k0..k0+15;
// lanes 16-31 row=lane-16, K = k0+16..k0+31.
__device__ __forceinline__ v16h load_a_h(const _Float16* __restrict__ X, int ldk,
                                         int row0, int k0, int lane) {
  const _Float16* p = X + (size_t)(row0 + (lane & 15)) * ldk + k0 + ((lane >> 4) << 4);
  v8h lo = *(const v8h*)p;          // 16B aligned
  v8h hi = *(const v8h*)(p + 8);    // 16B aligned
  v16h a;
#pragma unroll
  for (int i = 0; i < 8; ++i) { a[i] = lo[i]; a[8 + i] = hi[i]; }
  return a;
}

// B tile 32x16 (KxN) from f16 TRANSPOSED weight Wt[N][K] (row-major, ld=ldk).
// Lanes 0-15: col=lane, K=k0..k0+15 ; lanes 16-31: col=lane-16, K=k0+16..k0+31.
// Each lane's 16 K-values are contiguous -> one 32B load (2x b128).
__device__ __forceinline__ v16h load_b_h(const _Float16* __restrict__ Wt, int ldk,
                                         int k0, int n0, int lane) {
  const _Float16* p = Wt + (size_t)(n0 + (lane & 15)) * ldk + k0 + ((lane >> 4) << 4);
  return *(const v16h*)p;           // 32B aligned
}

__device__ __forceinline__ v8f wmma_f16(v16h a, v16h b, v8f c) {
  return __builtin_amdgcn_wmma_f32_16x16x32_f16(false, a, false, b,
                                                (short)0, c, false, false);
}

// Ordered-uint encode for float atomic max.
__device__ __forceinline__ unsigned enc_f(float f) {
  unsigned u = __float_as_uint(f);
  return (u & 0x80000000u) ? ~u : (u | 0x80000000u);
}
__device__ __forceinline__ float dec_f(unsigned u) {
  return (u & 0x80000000u) ? __uint_as_float(u & 0x7FFFFFFFu)
                           : __uint_as_float(~u);
}
__device__ __forceinline__ float lrelu(float v) { return v > 0.f ? v : SLOPE * v; }

// ---------------- Prep kernels (f32 -> f16, transpose weights) ----------------

// Transposed f16 copies: Wenc_t[64][32], Wgat_t[256][64], Wres_t[64][64].
__global__ __launch_bounds__(256) void k_cvt_w(const float* __restrict__ Wenc,
                                               const float* __restrict__ Wgat,
                                               const float* __restrict__ Wres,
                                               _Float16* __restrict__ wenc_t,
                                               _Float16* __restrict__ wgat_t,
                                               _Float16* __restrict__ wres_t) {
  int tid = blockIdx.x * blockDim.x + threadIdx.x;
  if (tid < HID * DIN) {                       // 2048
    int c = tid / DIN, k = tid % DIN;
    wenc_t[tid] = (_Float16)Wenc[k * HID + c];
    return;
  }
  tid -= HID * DIN;
  if (tid < HEADS * HID * HID) {               // 16384
    int c = tid / HID, k = tid % HID;
    wgat_t[tid] = (_Float16)Wgat[k * (HEADS * HID) + c];
    return;
  }
  tid -= HEADS * HID * HID;
  if (tid < HID * HID) {                       // 4096
    int c = tid / HID, k = tid % HID;
    wres_t[tid] = (_Float16)Wres[k * HID + c];
  }
}

// h[B,N,DIN] -> hh[r][DIN] f16 with r = n*BATCH + b
__global__ __launch_bounds__(256) void k_cvt_h(const float* __restrict__ h,
                                               _Float16* __restrict__ hh) {
  int tid = blockIdx.x * blockDim.x + threadIdx.x;
  if (tid >= MROWS * DIN) return;
  int r = tid / DIN, k = tid % DIN;
  int n = r / BATCH, b = r % BATCH;
  hh[tid] = (_Float16)h[((size_t)b * N_NODES + n) * DIN + k];
}

// f32 buffer -> f16 shadow (vectorized, count multiple of 8)
__global__ __launch_bounds__(256) void k_cvt_x(const float* __restrict__ src,
                                               _Float16* __restrict__ dst,
                                               int nElem) {
  int t = (blockIdx.x * blockDim.x + threadIdx.x) * 8;
  if (t >= nElem) return;
  const float4* s = (const float4*)(src + t);
  float4 a = s[0], b = s[1];
  v8h o;
  o[0] = (_Float16)a.x; o[1] = (_Float16)a.y;
  o[2] = (_Float16)a.z; o[3] = (_Float16)a.w;
  o[4] = (_Float16)b.x; o[5] = (_Float16)b.y;
  o[6] = (_Float16)b.z; o[7] = (_Float16)b.w;
  *(v8h*)(dst + t) = o;
}

// ---------------- GEMM kernels ----------------

// x[r, :HID] = hh[r,:DIN] @ Wenc + benc
__global__ __launch_bounds__(32) void k_enc(const _Float16* __restrict__ hh,
                                            const _Float16* __restrict__ wenc_t,
                                            const float* __restrict__ benc,
                                            float* __restrict__ x) {
  int lane = threadIdx.x;
  int row0 = blockIdx.x * 16;
  v16h a = load_a_h(hh, DIN, row0, 0, lane);
  v8f acc[4] = {};
#pragma unroll
  for (int j = 0; j < 4; ++j) {
    v16h bt = load_b_h(wenc_t, DIN, 0, j * 16, lane);
    acc[j] = wmma_f16(a, bt, acc[j]);
  }
  int cl = lane & 15;
  int rb = row0 + ((lane >> 4) << 3);
#pragma unroll
  for (int j = 0; j < 4; ++j)
#pragma unroll
    for (int i = 0; i < 8; ++i) {
      int col = j * 16 + cl;
      x[(size_t)(rb + i) * HID + col] = acc[j][i] + benc[col];
    }
}

// ft[r, 0:256] = x[r,:] @ Wgat (bias applied post-aggregation)
__global__ __launch_bounds__(32) void k_gatproj(const _Float16* __restrict__ xh,
                                                const _Float16* __restrict__ wgat_t,
                                                float* __restrict__ ft) {
  int lane = threadIdx.x;
  int row0 = blockIdx.x * 16;
  int n0   = blockIdx.y * 64;
  v8f acc[4] = {};
#pragma unroll
  for (int k0 = 0; k0 < HID; k0 += 32) {
    v16h a = load_a_h(xh, HID, row0, k0, lane);
#pragma unroll
    for (int j = 0; j < 4; ++j) {
      v16h bt = load_b_h(wgat_t, HID, k0, n0 + j * 16, lane);
      acc[j] = wmma_f16(a, bt, acc[j]);
    }
  }
  int cl = lane & 15;
  int rb = row0 + ((lane >> 4) << 3);
#pragma unroll
  for (int j = 0; j < 4; ++j)
#pragma unroll
    for (int i = 0; i < 8; ++i) {
      int col = n0 + j * 16 + cl;
      ft[(size_t)(rb + i) * (HEADS * HID) + col] = acc[j][i];
    }
}

// xnew = elu(agg_mean + mean_h(b_gat) + x @ Wres + bres)
__global__ __launch_bounds__(32) void k_combine(const _Float16* __restrict__ xh,
                                                const _Float16* __restrict__ wres_t,
                                                const float* __restrict__ bres,
                                                const float* __restrict__ bgat,
                                                const float* __restrict__ agg,
                                                float* __restrict__ xnew) {
  int lane = threadIdx.x;
  int row0 = blockIdx.x * 16;
  v8f acc[4] = {};
#pragma unroll
  for (int k0 = 0; k0 < HID; k0 += 32) {
    v16h a = load_a_h(xh, HID, row0, k0, lane);
#pragma unroll
    for (int j = 0; j < 4; ++j) {
      v16h bt = load_b_h(wres_t, HID, k0, j * 16, lane);
      acc[j] = wmma_f16(a, bt, acc[j]);
    }
  }
  int cl = lane & 15;
  int rb = row0 + ((lane >> 4) << 3);
#pragma unroll
  for (int j = 0; j < 4; ++j)
#pragma unroll
    for (int i = 0; i < 8; ++i) {
      int col = j * 16 + cl;
      float gb = 0.25f * (bgat[col] + bgat[HID + col] +
                          bgat[2 * HID + col] + bgat[3 * HID + col]);
      float v = acc[j][i] + bres[col] + gb +
                agg[(size_t)(rb + i) * HID + col];
      v = v > 0.f ? v : (__expf(v) - 1.f);
      xnew[(size_t)(rb + i) * HID + col] = v;
    }
}

// ---------------- Attention / edge kernels ----------------

// el[r,h] = ft[r,h,:] . a_l[h,:] ; er likewise (vectorized float4)
__global__ __launch_bounds__(256) void k_el_er(const float* __restrict__ ft,
                                               const float* __restrict__ al,
                                               const float* __restrict__ ar,
                                               float* __restrict__ el,
                                               float* __restrict__ er) {
  int tid = blockIdx.x * blockDim.x + threadIdx.x;
  if (tid >= MROWS * HEADS) return;
  int hh = tid & (HEADS - 1);
  const float4* f = (const float4*)(ft + (size_t)tid * HID);
  const float4* l = (const float4*)(al + hh * HID);
  const float4* r = (const float4*)(ar + hh * HID);
  float sl = 0.f, sr = 0.f;
#pragma unroll
  for (int d = 0; d < HID / 4; ++d) {
    float4 v = f[d], lv = l[d], rv = r[d];
    sl += v.x * lv.x + v.y * lv.y + v.z * lv.z + v.w * lv.w;
    sr += v.x * rv.x + v.y * rv.y + v.z * rv.z + v.w * rv.w;
  }
  el[tid] = sl;
  er[tid] = sr;
}

__global__ __launch_bounds__(256) void k_init(unsigned* __restrict__ mx,
                                              float* __restrict__ z,
                                              float* __restrict__ agg) {
  int tid = blockIdx.x * blockDim.x + threadIdx.x;
  if (tid < MROWS * HID) agg[tid] = 0.f;
  if (tid < MROWS * HEADS) { mx[tid] = 0u; z[tid] = 0.f; }
}

// pass 1: segment max of leaky(el[src]+er[dst]) over dst
__global__ __launch_bounds__(256) void k_edge_max(const int* __restrict__ ei,
                                                  const float* __restrict__ el,
                                                  const float* __restrict__ er,
                                                  unsigned* __restrict__ mx) {
  int tid = blockIdx.x * blockDim.x + threadIdx.x;
  if (tid >= N_EDGES * BATCH * HEADS) return;
  int e  = tid >> 3;
  int bh = tid & 7;
  int b  = bh >> 2, hh = bh & 3;
  int s = ei[e], t = ei[N_EDGES + e];
  float v = lrelu(el[(s * BATCH + b) * HEADS + hh] +
                  er[(t * BATCH + b) * HEADS + hh]);
  atomicMax(&mx[(t * BATCH + b) * HEADS + hh], enc_f(v));
}

// pass 2: w = exp(e - max), segment sum, cache w per edge
__global__ __launch_bounds__(256) void k_edge_exp(const int* __restrict__ ei,
                                                  const float* __restrict__ el,
                                                  const float* __restrict__ er,
                                                  const unsigned* __restrict__ mx,
                                                  float* __restrict__ z,
                                                  float* __restrict__ ew) {
  int tid = blockIdx.x * blockDim.x + threadIdx.x;
  if (tid >= N_EDGES * BATCH * HEADS) return;
  int e  = tid >> 3;
  int bh = tid & 7;
  int b  = bh >> 2, hh = bh & 3;
  int s = ei[e], t = ei[N_EDGES + e];
  int ti = (t * BATCH + b) * HEADS + hh;
  float v = lrelu(el[(s * BATCH + b) * HEADS + hh] + er[ti]);
  float w = __expf(v - dec_f(mx[ti]));
  ew[tid] = w;
  atomicAdd(&z[ti], w);
}

// pass 3: agg[dst,b,d] += (1/HEADS) * sum_h alpha * ft[src,b,h,d]
__global__ __launch_bounds__(256) void k_edge_agg(const int* __restrict__ ei,
                                                  const float* __restrict__ ft,
                                                  const float* __restrict__ ew,
                                                  const float* __restrict__ z,
                                                  float* __restrict__ agg) {
  int tid = blockIdx.x * blockDim.x + threadIdx.x;
  if (tid >= N_EDGES * BATCH * HID) return;
  int e   = tid >> 7;
  int rem = tid & 127;
  int b   = rem >> 6, d = rem & 63;
  int s = ei[e], t = ei[N_EDGES + e];
  int eb = (e * BATCH + b) * HEADS;
  int tb = (t * BATCH + b) * HEADS;
  const float* fs = ft + (size_t)(s * BATCH + b) * (HEADS * HID) + d;
  float acc = 0.f;
#pragma unroll
  for (int hh = 0; hh < HEADS; ++hh) {
    float alpha = ew[eb + hh] / z[tb + hh];
    acc += alpha * fs[hh * HID];
  }
  atomicAdd(&agg[(size_t)(t * BATCH + b) * HID + d], acc * (1.f / HEADS));
}

// out[b,n,o] = x[r,:] @ Wdec + bdec
__global__ __launch_bounds__(256) void k_dec(const float* __restrict__ x,
                                             const float* __restrict__ Wdec,
                                             const float* __restrict__ bdec,
                                             float* __restrict__ out) {
  int r = blockIdx.x * blockDim.x + threadIdx.x;
  if (r >= MROWS) return;
  int n = r / BATCH, b = r % BATCH;
  float acc[DOUT];
#pragma unroll
  for (int o = 0; o < DOUT; ++o) acc[o] = bdec[o];
  const float* xr = x + (size_t)r * HID;
  for (int k = 0; k < HID; ++k) {
    float xv = xr[k];
#pragma unroll
    for (int o = 0; o < DOUT; ++o) acc[o] += xv * Wdec[k * DOUT + o];
  }
  float* op = out + ((size_t)b * N_NODES + n) * DOUT;
#pragma unroll
  for (int o = 0; o < DOUT; ++o) op[o] = acc[o];
}

// ---------------- Host launcher ----------------

extern "C" void kernel_launch(void* const* d_in, const int* in_sizes, int n_in,
                              void* d_out, int out_size, void* d_ws, size_t ws_size,
                              hipStream_t stream) {
  const float* h    = (const float*)d_in[0];
  const int*   ei   = (const int*)d_in[1];
  const float* Wenc = (const float*)d_in[2];
  const float* benc = (const float*)d_in[3];
  const float* Wgat = (const float*)d_in[4];
  const float* a_l  = (const float*)d_in[5];
  const float* a_r  = (const float*)d_in[6];
  const float* bgat = (const float*)d_in[7];
  const float* Wres = (const float*)d_in[8];
  const float* bres = (const float*)d_in[9];
  const float* Wdec = (const float*)d_in[10];
  const float* bdec = (const float*)d_in[11];
  float* out = (float*)d_out;

  // f32 region
  float*    x0  = (float*)d_ws;                            // MROWS*HID
  float*    x1  = x0 + (size_t)MROWS * HID;                // MROWS*HID
  float*    ft  = x1 + (size_t)MROWS * HID;                // MROWS*HEADS*HID
  float*    el  = ft + (size_t)MROWS * HEADS * HID;        // MROWS*HEADS
  float*    er  = el + (size_t)MROWS * HEADS;              // MROWS*HEADS
  unsigned* mx  = (unsigned*)(er + (size_t)MROWS * HEADS); // MROWS*HEADS
  float*    z   = (float*)(mx + (size_t)MROWS * HEADS);    // MROWS*HEADS
  float*    ew  = z + (size_t)MROWS * HEADS;               // E*BATCH*HEADS
  float*    agg = ew + (size_t)N_EDGES * BATCH * HEADS;    // MROWS*HID
  // f16 region (32B aligned by construction)
  _Float16* xh     = (_Float16*)(agg + (size_t)MROWS * HID); // MROWS*HID
  _Float16* hh     = xh + (size_t)MROWS * HID;               // MROWS*DIN
  _Float16* wenc_t = hh + (size_t)MROWS * DIN;               // HID*DIN
  _Float16* wgat_t = wenc_t + HID * DIN;                     // (HEADS*HID)*HID
  _Float16* wres_t = wgat_t + HEADS * HID * HID;             // HID*HID

  const int WCNT = HID * DIN + HEADS * HID * HID + HID * HID;
  k_cvt_w<<<(WCNT + 255) / 256, 256, 0, stream>>>(Wenc, Wgat, Wres,
                                                  wenc_t, wgat_t, wres_t);
  k_cvt_h<<<(MROWS * DIN + 255) / 256, 256, 0, stream>>>(h, hh);

  k_enc<<<MROWS / 16, 32, 0, stream>>>(hh, wenc_t, benc, x0);
  const int XN = MROWS * HID;
  k_cvt_x<<<(XN / 8 + 255) / 256, 256, 0, stream>>>(x0, xh, XN);

  float* cur = x0;
  float* nxt = x1;
  for (int it = 0; it < 2; ++it) {
    k_init<<<(XN + 255) / 256, 256, 0, stream>>>(mx, z, agg);
    k_gatproj<<<dim3(MROWS / 16, HEADS), 32, 0, stream>>>(xh, wgat_t, ft);
    k_el_er<<<(MROWS * HEADS + 255) / 256, 256, 0, stream>>>(ft, a_l, a_r, el, er);
    int ne = N_EDGES * BATCH * HEADS;
    k_edge_max<<<(ne + 255) / 256, 256, 0, stream>>>(ei, el, er, mx);
    k_edge_exp<<<(ne + 255) / 256, 256, 0, stream>>>(ei, el, er, mx, z, ew);
    int na = N_EDGES * BATCH * HID;
    k_edge_agg<<<(na + 255) / 256, 256, 0, stream>>>(ei, ft, ew, z, agg);
    k_combine<<<MROWS / 16, 32, 0, stream>>>(xh, wres_t, bres, bgat, agg, nxt);
    k_cvt_x<<<(XN / 8 + 255) / 256, 256, 0, stream>>>(nxt, xh, XN);
    float* t = cur; cur = nxt; nxt = t;
  }
  k_dec<<<(MROWS + 255) / 256, 256, 0, stream>>>(cur, Wdec, bdec, out);
}
